// Attention_13280038880137
// MI455X (gfx1250) — matrix operand order repaired and checked by reading, not compiled, not run on hardware
//
#include <hip/hip_runtime.h>
#include <hip/hip_bf16.h>

// ---------------- problem constants ----------------
#define DIMC   512
#define NTOK   8192
#define BATCH  4
#define NHEAD  8
#define DHEAD  64
#define TRIPLE 1536                  // 3 * 512
#define ROWS   (BATCH * NTOK)        // 32768 tokens total
#define SCALE  0.125f                // 64^-0.5
#define EPS    1e-5f

typedef __attribute__((ext_vector_type(16))) _Float16 v16h;
typedef __attribute__((ext_vector_type(8)))  _Float16 v8h;
typedef __attribute__((ext_vector_type(8)))  float    v8f;
typedef __attribute__((ext_vector_type(4)))  float    v4f;

// ---------------- workspace layout (bytes) ----------------
// all offsets multiples of 256
#define OFF_HN   ((size_t)0)            // f16 ROWS x 512 (layernorm out; reused as attn out)
#define OFF_Q    ((size_t)33554432)     // f16 ROWS x 512 (q, later q-weights in place)
#define OFF_KT   ((size_t)67108864)     // f16 (B*H, 64, NTOK)  k^T, softmaxed in place
#define OFF_VT   ((size_t)100663296)    // f16 (B*H, 64, NTOK)  v^T
#define OFF_WQKV ((size_t)134217728)    // f16 1536 x 512 (w_qkv transposed)
#define OFF_WOUT ((size_t)135790592)    // f16 512 x 512 (w_out transposed)
#define OFF_CTX  ((size_t)136314880)    // f32 (B*H) x 64 x 64
#define OFF_CTXT ((size_t)136839168)    // f16 (B*H) x 64(e) x 64(d)  (transposed ctx)
#define OFF_CMAX ((size_t)137101312)    // f32 B*512
#define OFF_CSUM ((size_t)137109504)    // f32 B*512

// ---------------- WMMA fragment helpers ----------------
// A fragment 16x32 (MxK) from row-major f16, base pre-offset to (row0,col0).
__device__ __forceinline__ v16h load_a_f16(const _Float16* base, long ld, int lane) {
  int r  = lane & 15;
  int h8 = (lane >> 4) << 3;                 // 0 or 8
  const _Float16* p = base + (long)r * ld + h8;
  v8h lo = *(const v8h*)p;                   // K = h8 .. h8+7
  v8h hi = *(const v8h*)(p + 16);            // K = 16+h8 .. 16+h8+7
  v16h out;
#pragma unroll
  for (int i = 0; i < 8; ++i) { out[i] = lo[i]; out[i + 8] = hi[i]; }
  return out;
}

// B fragment 32x16 (KxN) from B^T row-major (N x K), base pre-offset to (n0,k0).
__device__ __forceinline__ v16h load_b_f16(const _Float16* baseT, long ld, int lane) {
  int n  = lane & 15;
  int kb = (lane >> 4) << 3;                 // 0 or 8
  const _Float16* p = baseT + (long)n * ld + kb;
  v8h lo = *(const v8h*)p;
  v8h hi = *(const v8h*)(p + 16);
  v16h out;
#pragma unroll
  for (int i = 0; i < 8; ++i) { out[i] = lo[i]; out[i + 8] = hi[i]; }
  return out;
}

__device__ __forceinline__ v8f wmma32(v16h a, v16h b, v8f c) {
  return __builtin_amdgcn_wmma_f32_16x16x32_f16(false, a, false, b, (short)0, c,
                                                false, false);
}

// ---------------- kernel 1: weight transpose + f16 convert ----------------
__global__ __launch_bounds__(256) void wconv_kernel(const float* w_qkv, const float* w_out,
                                                    _Float16* wq, _Float16* wo) {
  int idx = blockIdx.x * 256 + threadIdx.x;          // 0 .. 1048575
  if (idx < TRIPLE * DIMC) {
    int o = idx / DIMC, i = idx % DIMC;              // wq[o][i] = w_qkv[i][o]
    wq[idx] = (_Float16)w_qkv[(long)i * TRIPLE + o];
  } else {
    int j = idx - TRIPLE * DIMC;
    int o = j / DIMC, i = j % DIMC;                  // wo[o][i] = w_out[i][o]
    wo[j] = (_Float16)w_out[(long)i * DIMC + o];
  }
}

// ---------------- kernel 2: LayerNorm (transposing read via LDS) ----------------
__global__ __launch_bounds__(256) void ln_kernel(const float* x, const float* ln_w,
                                                 const float* ln_b, _Float16* hn) {
  __shared__ float tile[32][513];                    // 32 tokens x 512 dims (+pad)
  int b  = blockIdx.y;
  int n0 = blockIdx.x * 32;
  int tx = threadIdx.x & 31;                         // token in tile
  int ty = threadIdx.x >> 5;                         // 0..7
  const float* xb = x + (long)b * DIMC * NTOK;
#pragma unroll 4
  for (int d0 = 0; d0 < DIMC; d0 += 8)
    tile[tx][d0 + ty] = xb[(long)(d0 + ty) * NTOK + n0 + tx];   // coalesced along n
  __syncthreads();
  int lane = tx, wv = ty;
  for (int j = 0; j < 4; ++j) {
    int t = wv * 4 + j;
    float s = 0.f, ss = 0.f;
#pragma unroll
    for (int i = 0; i < 16; ++i) { float v = tile[t][i * 32 + lane]; s += v; ss += v * v; }
#pragma unroll
    for (int off = 16; off; off >>= 1) { s += __shfl_xor(s, off, 32); ss += __shfl_xor(ss, off, 32); }
    float mu = s * (1.f / DIMC);
    float rs = rsqrtf(ss * (1.f / DIMC) - mu * mu + EPS);
    long gtok = (long)b * NTOK + n0 + t;
#pragma unroll
    for (int i = 0; i < 16; ++i) {
      int d = i * 32 + lane;
      hn[gtok * DIMC + d] = (_Float16)((tile[t][d] - mu) * rs * ln_w[d] + ln_b[d]);
    }
  }
}

// ---------------- kernel 3: QKV GEMM (ROWS x 1536 = hn @ Wqkv) ----------------
__global__ __launch_bounds__(256) void gemm_qkv_kernel(const _Float16* hn, const _Float16* wt,
                                                       _Float16* q, _Float16* kT, _Float16* vT) {
  int lane = threadIdx.x & 31, wv = threadIdx.x >> 5;
  int wm = wv >> 2, wn = wv & 3;
  long m0 = (long)blockIdx.y * 128 + wm * 64;
  int  n0 = blockIdx.x * 256 + wn * 64;
  v8f acc[4][4];
#pragma unroll
  for (int mi = 0; mi < 4; ++mi)
#pragma unroll
    for (int ni = 0; ni < 4; ++ni) acc[mi][ni] = (v8f){};
  for (int k0 = 0; k0 < DIMC; k0 += 32) {
    if (k0 + 64 < DIMC) {                             // gfx1250 global_prefetch two k-steps ahead
      __builtin_prefetch(hn + m0 * DIMC + k0 + 64, 0, 3);
      __builtin_prefetch(wt + (long)n0 * DIMC + k0 + 64, 0, 3);
    }
    v16h a[4];
#pragma unroll
    for (int mi = 0; mi < 4; ++mi)
      a[mi] = load_a_f16(hn + (m0 + mi * 16) * DIMC + k0, DIMC, lane);
#pragma unroll
    for (int ni = 0; ni < 4; ++ni) {
      v16h bf = load_b_f16(wt + (long)(n0 + ni * 16) * DIMC + k0, DIMC, lane);
#pragma unroll
      for (int mi = 0; mi < 4; ++mi) acc[mi][ni] = wmma32(a[mi], bf, acc[mi][ni]);
    }
  }
  int nc = lane & 15, mr = (lane >> 4) * 8;
#pragma unroll
  for (int mi = 0; mi < 4; ++mi) {
    long mb  = m0 + mi * 16 + mr;                     // 8 consecutive tokens per lane
    long bb  = mb >> 13, tokb = mb & (NTOK - 1);
#pragma unroll
    for (int ni = 0; ni < 4; ++ni) {
      int cbase = n0 + ni * 16;                       // tile stays inside one q/k/v region
      int c = cbase + nc;
      if (cbase < DIMC) {                             // q: row-major, scalar stores
#pragma unroll
        for (int r = 0; r < 8; ++r) q[(mb + r) * DIMC + c] = (_Float16)acc[mi][ni][r];
      } else if (cbase < 2 * DIMC) {                  // k^T: token-contiguous -> b128 store
        int d = c - DIMC, h = d >> 6, dd = d & 63;
        v8h pk;
#pragma unroll
        for (int r = 0; r < 8; ++r) pk[r] = (_Float16)acc[mi][ni][r];
        *(v8h*)(kT + (((bb * NHEAD + h) << 6) + dd) * (long)NTOK + tokb) = pk;
      } else {                                        // v^T
        int e = c - 2 * DIMC, h = e >> 6, ee = e & 63;
        v8h pk;
#pragma unroll
        for (int r = 0; r < 8; ++r) pk[r] = (_Float16)acc[mi][ni][r];
        *(v8h*)(vT + (((bb * NHEAD + h) << 6) + ee) * (long)NTOK + tokb) = pk;
      }
    }
  }
}

// ---------------- kernel 4: k softmax over head-dim, in place on kT ----------------
__global__ __launch_bounds__(256) void ksoftmax_kernel(_Float16* kT) {
  int bh = blockIdx.y;
  long tok = (long)blockIdx.x * 256 + threadIdx.x;
  _Float16* base = kT + (long)bh * DHEAD * NTOK + tok;
  float v[DHEAD], m = -1e30f;
#pragma unroll
  for (int d = 0; d < DHEAD; ++d) { v[d] = (float)base[(long)d * NTOK]; m = fmaxf(m, v[d]); }
  float s = 0.f;
#pragma unroll
  for (int d = 0; d < DHEAD; ++d) { v[d] = __expf(v[d] - m); s += v[d]; }
  float inv = 1.f / s;
#pragma unroll
  for (int d = 0; d < DHEAD; ++d) base[(long)d * NTOK] = (_Float16)(v[d] * inv);
}

// ---------------- kernel 5: q column stats (softmax over tokens) ----------------
// block = 16 cols x 16 token-strides: 32B-contiguous reads per half-wave
__global__ __launch_bounds__(256) void colstats_kernel(const _Float16* q, float* cmax, float* csum) {
  __shared__ float buf[16][17];
  int col0 = blockIdx.x * 16, b = blockIdx.y;
  int tx = threadIdx.x & 15;                          // column
  int ty = threadIdx.x >> 4;                          // token stride 0..15
  const _Float16* base = q + ((long)b * NTOK) * DIMC + col0 + tx;
  float m = -1e30f;
  for (int n = ty; n < NTOK; n += 16) m = fmaxf(m, (float)base[(long)n * DIMC] * SCALE);
  buf[ty][tx] = m; __syncthreads();
  for (int off = 8; off; off >>= 1) {
    if (ty < off) buf[ty][tx] = fmaxf(buf[ty][tx], buf[ty + off][tx]);
    __syncthreads();
  }
  float M = buf[0][tx];
  __syncthreads();
  float s = 0.f;
  for (int n = ty; n < NTOK; n += 16) s += __expf((float)base[(long)n * DIMC] * SCALE - M);
  buf[ty][tx] = s; __syncthreads();
  for (int off = 8; off; off >>= 1) {
    if (ty < off) buf[ty][tx] += buf[ty + off][tx];
    __syncthreads();
  }
  if (ty == 0) { cmax[b * DIMC + col0 + tx] = M; csum[b * DIMC + col0 + tx] = buf[0][tx]; }
}

// ---------------- kernel 6: q -> softmax weights, in place ----------------
__global__ __launch_bounds__(256) void qweight_kernel(_Float16* q, const float* cmax, const float* csum) {
  long idx = (long)blockIdx.x * 256 + threadIdx.x;   // over ROWS*512
  int col = (int)(idx & (DIMC - 1));
  long row = idx >> 9;
  int b = (int)(row >> 13);
  float w = __expf((float)q[idx] * SCALE - cmax[b * DIMC + col]) * (1.f / csum[b * DIMC + col]);
  q[idx] = (_Float16)w;
}

// ---------------- kernel 7: zero context accumulator ----------------
__global__ __launch_bounds__(256) void zero_kernel(float* p) {
  p[blockIdx.x * 256 + threadIdx.x] = 0.f;
}

// ---------------- kernel 8: context = k_sm^T @ v  (64x64 per head, split-K) -------
__global__ __launch_bounds__(32) void ctx_gemm_kernel(const _Float16* kT, const _Float16* vT,
                                                      float* ctx) {
  int bh = blockIdx.x, kc = blockIdx.y, lane = threadIdx.x & 31;
  const _Float16* A  = kT + (long)bh * DHEAD * NTOK;   // (64 d) x (NTOK) row-major
  const _Float16* Bt = vT + (long)bh * DHEAD * NTOK;   // B^T: (64 e) x (NTOK)
  v8f acc[4][4];
#pragma unroll
  for (int mi = 0; mi < 4; ++mi)
#pragma unroll
    for (int ni = 0; ni < 4; ++ni) acc[mi][ni] = (v8f){};
  long kbase = (long)kc * (NTOK / 16);
  for (int kk = 0; kk < NTOK / 16; kk += 32) {
    long k0 = kbase + kk;
    v16h a[4];
#pragma unroll
    for (int mi = 0; mi < 4; ++mi)
      a[mi] = load_a_f16(A + (long)(mi * 16) * NTOK + k0, NTOK, lane);
#pragma unroll
    for (int ni = 0; ni < 4; ++ni) {
      v16h bf = load_b_f16(Bt + (long)(ni * 16) * NTOK + k0, NTOK, lane);
#pragma unroll
      for (int mi = 0; mi < 4; ++mi) acc[mi][ni] = wmma32(a[mi], bf, acc[mi][ni]);
    }
  }
  float* C = ctx + (long)bh * DHEAD * DHEAD;
  int nc = lane & 15, mr = (lane >> 4) * 8;
#pragma unroll
  for (int mi = 0; mi < 4; ++mi)
#pragma unroll
    for (int ni = 0; ni < 4; ++ni)
#pragma unroll
      for (int r = 0; r < 8; ++r)
        atomicAdd(&C[(mi * 16 + mr + r) * DHEAD + ni * 16 + nc], acc[mi][ni][r]);
}

// ---------------- kernel 9: ctx -> f16 transposed (e x d) ----------------
__global__ __launch_bounds__(256) void ctxcvt_kernel(const float* ctx, _Float16* ctxhT) {
  int idx = blockIdx.x * 256 + threadIdx.x;          // 0..131071
  int bh = idx >> 12, r = idx & 4095, d = r >> 6, e = r & 63;
  ctxhT[(bh << 12) + (e << 6) + d] = (_Float16)ctx[idx];
}

// ---------------- kernel 10: attn = q_w @ ctx (per head, K=64) ----------------
__global__ __launch_bounds__(256) void attn_gemm_kernel(const _Float16* q, const _Float16* ctxhT,
                                                        _Float16* attn) {
  int bh = blockIdx.x, b = bh >> 3, h = bh & 7;
  int lane = threadIdx.x & 31, wv = threadIdx.x >> 5;
  long tok0 = (long)blockIdx.y * 512 + wv * 64;
  const _Float16* A  = q + ((long)b * NTOK + tok0) * DIMC + h * DHEAD;  // ld 512
  const _Float16* Bt = ctxhT + (long)bh * DHEAD * DHEAD;                // (e x d), ld 64
  v8f acc[4][4];
#pragma unroll
  for (int mi = 0; mi < 4; ++mi)
#pragma unroll
    for (int ni = 0; ni < 4; ++ni) acc[mi][ni] = (v8f){};
#pragma unroll
  for (int k0 = 0; k0 < DHEAD; k0 += 32) {
    v16h a[4];
#pragma unroll
    for (int mi = 0; mi < 4; ++mi)
      a[mi] = load_a_f16(A + (long)(mi * 16) * DIMC + k0, DIMC, lane);
#pragma unroll
    for (int ni = 0; ni < 4; ++ni) {
      v16h bf = load_b_f16(Bt + (long)(ni * 16) * DHEAD + k0, DHEAD, lane);
#pragma unroll
      for (int mi = 0; mi < 4; ++mi) acc[mi][ni] = wmma32(a[mi], bf, acc[mi][ni]);
    }
  }
  int nc = lane & 15, mr = (lane >> 4) * 8;
#pragma unroll
  for (int mi = 0; mi < 4; ++mi)
#pragma unroll
    for (int ni = 0; ni < 4; ++ni)
#pragma unroll
      for (int r = 0; r < 8; ++r) {
        long row = (long)b * NTOK + tok0 + mi * 16 + mr + r;
        attn[row * DIMC + h * DHEAD + ni * 16 + nc] = (_Float16)acc[mi][ni][r];
      }
}

// ---------------- kernel 11: out = attn @ Wout + b_out, transposed store ----------
__global__ __launch_bounds__(256) void out_gemm_kernel(const _Float16* attn, const _Float16* wt,
                                                       const float* b_out, float* out) {
  int lane = threadIdx.x & 31, wv = threadIdx.x >> 5;
  int wm = wv >> 2, wn = wv & 3;
  long m0 = (long)blockIdx.y * 128 + wm * 64;
  int  n0 = blockIdx.x * 256 + wn * 64;
  v8f acc[4][4];
#pragma unroll
  for (int mi = 0; mi < 4; ++mi)
#pragma unroll
    for (int ni = 0; ni < 4; ++ni) acc[mi][ni] = (v8f){};
  for (int k0 = 0; k0 < DIMC; k0 += 32) {
    if (k0 + 64 < DIMC) {
      __builtin_prefetch(attn + m0 * DIMC + k0 + 64, 0, 3);
      __builtin_prefetch(wt + (long)n0 * DIMC + k0 + 64, 0, 3);
    }
    v16h a[4];
#pragma unroll
    for (int mi = 0; mi < 4; ++mi)
      a[mi] = load_a_f16(attn + (m0 + mi * 16) * DIMC + k0, DIMC, lane);
#pragma unroll
    for (int ni = 0; ni < 4; ++ni) {
      v16h bf = load_b_f16(wt + (long)(n0 + ni * 16) * DIMC + k0, DIMC, lane);
#pragma unroll
      for (int mi = 0; mi < 4; ++mi) acc[mi][ni] = wmma32(a[mi], bf, acc[mi][ni]);
    }
  }
  // D rows (VGPR r) are consecutive tokens; out is (B, 512dim, Ntok) -> vector stores
  int nc = lane & 15, mr = (lane >> 4) * 8;
#pragma unroll
  for (int mi = 0; mi < 4; ++mi) {
    long mb  = m0 + mi * 16 + mr;
    long bb  = mb >> 13, tokb = mb & (NTOK - 1);
#pragma unroll
    for (int ni = 0; ni < 4; ++ni) {
      int c = n0 + ni * 16 + nc;
      float bias = b_out[c];
      float* p = out + ((bb << 9) + c) * (long)NTOK + tokb;
      v4f lo, hi;
#pragma unroll
      for (int r = 0; r < 4; ++r) { lo[r] = acc[mi][ni][r] + bias; hi[r] = acc[mi][ni][4 + r] + bias; }
      *(v4f*)p       = lo;                            // global_store_b128
      *(v4f*)(p + 4) = hi;                            // global_store_b128
    }
  }
}

// ---------------- launcher ----------------
extern "C" void kernel_launch(void* const* d_in, const int* in_sizes, int n_in,
                              void* d_out, int out_size, void* d_ws, size_t ws_size,
                              hipStream_t stream) {
  const float* x     = (const float*)d_in[0];
  const float* ln_w  = (const float*)d_in[1];
  const float* ln_b  = (const float*)d_in[2];
  const float* w_qkv = (const float*)d_in[3];
  const float* w_out = (const float*)d_in[4];
  const float* b_out = (const float*)d_in[5];
  float* out = (float*)d_out;

  char* ws = (char*)d_ws;
  _Float16* hn    = (_Float16*)(ws + OFF_HN);   // reused as attn
  _Float16* q     = (_Float16*)(ws + OFF_Q);
  _Float16* kT    = (_Float16*)(ws + OFF_KT);
  _Float16* vT    = (_Float16*)(ws + OFF_VT);
  _Float16* wq    = (_Float16*)(ws + OFF_WQKV);
  _Float16* wo    = (_Float16*)(ws + OFF_WOUT);
  float*    ctx   = (float*)   (ws + OFF_CTX);
  _Float16* ctxhT = (_Float16*)(ws + OFF_CTXT);
  float*    cmax  = (float*)   (ws + OFF_CMAX);
  float*    csum  = (float*)   (ws + OFF_CSUM);

  // 1. weights -> f16 transposed
  wconv_kernel<<<dim3((TRIPLE * DIMC + DIMC * DIMC) / 256), 256, 0, stream>>>(w_qkv, w_out, wq, wo);
  // 2. layernorm (B,D,N) -> hn f16 (ROWS, 512)
  ln_kernel<<<dim3(NTOK / 32, BATCH), 256, 0, stream>>>(x, ln_w, ln_b, hn);
  // 3. qkv GEMM -> q (row-major), kT, vT (per-head transposed)
  gemm_qkv_kernel<<<dim3(TRIPLE / 256, ROWS / 128), 256, 0, stream>>>(hn, wq, q, kT, vT);
  // 4. k softmax over head-dim (in place on kT)
  ksoftmax_kernel<<<dim3(NTOK / 256, BATCH * NHEAD), 256, 0, stream>>>(kT);
  // 5. q column stats (softmax over tokens)
  colstats_kernel<<<dim3(DIMC / 16, BATCH), 256, 0, stream>>>(q, cmax, csum);
  // 6. q -> normalized weights (in place)
  qweight_kernel<<<dim3((int)((long)ROWS * DIMC / 256)), 256, 0, stream>>>(q, cmax, csum);
  // 7. zero ctx
  zero_kernel<<<dim3(BATCH * NHEAD * DHEAD * DHEAD / 256), 256, 0, stream>>>(ctx);
  // 8. context = k_sm^T @ v, split-K 16 with f32 atomics
  ctx_gemm_kernel<<<dim3(BATCH * NHEAD, 16), 32, 0, stream>>>(kT, vT, ctx);
  // 9. ctx -> f16 transposed (e x d)
  ctxcvt_kernel<<<dim3(BATCH * NHEAD * DHEAD * DHEAD / 256), 256, 0, stream>>>(ctx, ctxhT);
  // 10. attn = q_w @ ctx   (reuses hn region)
  attn_gemm_kernel<<<dim3(BATCH * NHEAD, NTOK / 512), 256, 0, stream>>>(q, ctxhT, hn);
  // 11. out projection + bias + transpose to (B, 512, N)
  out_gemm_kernel<<<dim3(DIMC / 256, ROWS / 128), 256, 0, stream>>>(hn, wo, b_out, out);
}
